// QuantizedLinear_21663815041806
// MI455X (gfx1250) — compile-verified, matrix operand order checked by reading, not run on hardware
//
#include <hip/hip_runtime.h>
#include <hip/hip_bf16.h>

// ---------------------------------------------------------------------------
// QuantizedLinear for MI455X (gfx1250):
//   out = x @ (qW * scale)^T + qb * bias_scale
// bf16 WMMA GEMM (int8-valued weights are exact in bf16; scale in fp32
// epilogue).  Double-buffered LDS pipeline fed by gfx1250 async global->LDS
// copies (ASYNCcnt) -- no VGPR staging, no spills; copies overlap the
// 16x v_wmma_f32_16x16x32_bf16 per stage.
// ---------------------------------------------------------------------------

typedef __attribute__((ext_vector_type(16))) __bf16 v16bf;
typedef __attribute__((ext_vector_type(8)))  __bf16 v8bf;
typedef __attribute__((ext_vector_type(8)))  float  v8f;
typedef __attribute__((ext_vector_type(4)))  int    v4i;

#if defined(__has_builtin)
#if __has_builtin(__builtin_amdgcn_global_load_async_to_lds_b128)
#define HAVE_ASYNC_LDS 1
#endif
#if __has_builtin(__builtin_amdgcn_s_wait_asynccnt)
#define WAIT_ASYNC() __builtin_amdgcn_s_wait_asynccnt(0)
#endif
#endif
#ifndef WAIT_ASYNC
#define WAIT_ASYNC() asm volatile("s_wait_asynccnt 0" ::: "memory")
#endif

__device__ __forceinline__ unsigned short bf16_rne(float f) {
  unsigned int u = __float_as_uint(f);
  unsigned int r = u + 0x7FFFu + ((u >> 16) & 1u);
  return (unsigned short)(r >> 16);
}

// ------------------------- conversion kernels ------------------------------

__global__ __launch_bounds__(256) void cvt_f32_to_bf16(
    const float* __restrict__ in, unsigned short* __restrict__ out, long long n) {
  long long i = (((long long)blockIdx.x * 256) + threadIdx.x) * 8;
  if (i + 8 > n) return;
  float4 a = *(const float4*)(in + i);
  float4 b = *(const float4*)(in + i + 4);
  uint4 p;
  p.x = (unsigned)bf16_rne(a.x) | ((unsigned)bf16_rne(a.y) << 16);
  p.y = (unsigned)bf16_rne(a.z) | ((unsigned)bf16_rne(a.w) << 16);
  p.z = (unsigned)bf16_rne(b.x) | ((unsigned)bf16_rne(b.y) << 16);
  p.w = (unsigned)bf16_rne(b.z) | ((unsigned)bf16_rne(b.w) << 16);
  *(uint4*)(out + i) = p;
}

__global__ __launch_bounds__(256) void cvt_i32_to_bf16(
    const int* __restrict__ in, unsigned short* __restrict__ out, long long n) {
  long long i = (((long long)blockIdx.x * 256) + threadIdx.x) * 8;
  if (i + 8 > n) return;
  int4 a = *(const int4*)(in + i);
  int4 b = *(const int4*)(in + i + 4);
  uint4 p;  // values in [-127,127] convert exactly
  p.x = (unsigned)bf16_rne((float)a.x) | ((unsigned)bf16_rne((float)a.y) << 16);
  p.y = (unsigned)bf16_rne((float)a.z) | ((unsigned)bf16_rne((float)a.w) << 16);
  p.z = (unsigned)bf16_rne((float)b.x) | ((unsigned)bf16_rne((float)b.y) << 16);
  p.w = (unsigned)bf16_rne((float)b.z) | ((unsigned)bf16_rne((float)b.w) << 16);
  *(uint4*)(out + i) = p;
}

// ------------------------------ WMMA GEMM ----------------------------------

#define BM 128
#define BN 128
#define BK 64   // one LDS stage = two 32-deep WMMA sub-steps

__global__ __launch_bounds__(256) void qlinear_wmma_bf16(
    const unsigned short* __restrict__ X,   // [M,K] bf16 bits, row-major
    const unsigned short* __restrict__ W,   // [N,K] bf16 bits, row-major (== B^T)
    const int*            __restrict__ qbias,      // [N]
    const float*          __restrict__ scale,      // [1]
    const float*          __restrict__ bias_scale, // [1]
    float*                __restrict__ out,        // [M,N]
    int M, int N, int K) {
  __shared__ __align__(16) unsigned short As[2][BM * BK];  // 2 x 16 KB
  __shared__ __align__(16) unsigned short Ws[2][BN * BK];  // 2 x 16 KB

  const int tid  = threadIdx.x;
  const int wave = tid >> 5;
  const int lane = tid & 31;
  const int wm   = wave >> 2;   // 0..1 : 64-row band
  const int wn   = wave & 3;    // 0..3 : 32-col band
  const int lr   = lane & 15;
  const int lh   = lane >> 4;

  const size_t bm = (size_t)blockIdx.y * BM;
  const size_t bn = (size_t)blockIdx.x * BN;

  v8f acc[4][2];
#pragma unroll
  for (int i = 0; i < 4; ++i)
#pragma unroll
    for (int j = 0; j < 2; ++j)
#pragma unroll
      for (int e = 0; e < 8; ++e) acc[i][j][e] = 0.0f;

  // 1024 16B-chunks per 128x64 tile -> 4 chunks/thread for each of X and W.
#ifdef HAVE_ASYNC_LDS
  auto stage_tile = [&](int kt, int buf) {
#pragma unroll
    for (int c = 0; c < 4; ++c) {
      const int ch  = tid + c * 256;
      const int r   = ch >> 3;
      const int col = (ch & 7) << 3;
      __builtin_amdgcn_global_load_async_to_lds_b128(
          (v4i*)(void*)(X + (bm + (size_t)r) * (size_t)K + kt + col),
          (v4i*)&As[buf][r * BK + col], 0, 0);
      __builtin_amdgcn_global_load_async_to_lds_b128(
          (v4i*)(void*)(W + (bn + (size_t)r) * (size_t)K + kt + col),
          (v4i*)&Ws[buf][r * BK + col], 0, 0);
    }
  };
#else
  auto stage_tile = [&](int kt, int buf) {
    uint4 ar[4], wr[4];
#pragma unroll
    for (int c = 0; c < 4; ++c) {
      const int ch  = tid + c * 256;
      const int r   = ch >> 3;
      const int col = (ch & 7) << 3;
      ar[c] = *(const uint4*)(X + (bm + (size_t)r) * (size_t)K + kt + col);
      wr[c] = *(const uint4*)(W + (bn + (size_t)r) * (size_t)K + kt + col);
    }
#pragma unroll
    for (int c = 0; c < 4; ++c) {
      const int ch  = tid + c * 256;
      const int r   = ch >> 3;
      const int col = (ch & 7) << 3;
      *(uint4*)(&As[buf][r * BK + col]) = ar[c];
      *(uint4*)(&Ws[buf][r * BK + col]) = wr[c];
    }
  };
#endif

  // ---- prologue: stage tile 0 and wait for it ----
  stage_tile(0, 0);
#ifdef HAVE_ASYNC_LDS
  WAIT_ASYNC();
#endif
  __syncthreads();

  int cur = 0;
  for (int kt = 0; kt < K; kt += BK) {
    const bool has_next = (kt + BK) < K;
    if (has_next) stage_tile(kt + BK, cur ^ 1);   // async: overlaps WMMAs
    if (kt + 2 * BK < K) {                        // nudge t+2 toward L2
      __builtin_prefetch(X + (bm + (size_t)(tid >> 3)) * (size_t)K + kt + 2 * BK, 0, 3);
      __builtin_prefetch(W + (bn + (size_t)(tid >> 3)) * (size_t)K + kt + 2 * BK, 0, 3);
    }

    // ---- compute on LDS[cur]: 2 k-substeps x (4x2) fragments = 16 WMMA ----
#pragma unroll
    for (int kk = 0; kk < BK; kk += 32) {
      v16bf a[4], b[2];
#pragma unroll
      for (int j = 0; j < 2; ++j) {
        const unsigned short* p =
            &Ws[cur][(wn * 32 + j * 16 + lr) * BK + kk + lh * 8];
        v8bf q0 = *(const v8bf*)p;
        v8bf q1 = *(const v8bf*)(p + 16);
        b[j] = __builtin_shufflevector(q0, q1, 0, 1, 2, 3, 4, 5, 6, 7,
                                       8, 9, 10, 11, 12, 13, 14, 15);
      }
#pragma unroll
      for (int i = 0; i < 4; ++i) {
        const unsigned short* p =
            &As[cur][(wm * 64 + i * 16 + lr) * BK + kk + lh * 8];
        v8bf q0 = *(const v8bf*)p;
        v8bf q1 = *(const v8bf*)(p + 16);
        a[i] = __builtin_shufflevector(q0, q1, 0, 1, 2, 3, 4, 5, 6, 7,
                                       8, 9, 10, 11, 12, 13, 14, 15);
      }
#pragma unroll
      for (int i = 0; i < 4; ++i)
#pragma unroll
        for (int j = 0; j < 2; ++j)
          acc[i][j] = __builtin_amdgcn_wmma_f32_16x16x32_bf16(
              false, a[i], false, b[j], (short)0, acc[i][j], false, false);
    }

#ifdef HAVE_ASYNC_LDS
    if (has_next) WAIT_ASYNC();   // own async copies into LDS[cur^1] done
#endif
    __syncthreads();              // publish all waves' copies
    cur ^= 1;
  }

  // ---- fp32 epilogue: dequant scale + bias ----
  const float s  = scale[0];
  const float bs = bias_scale[0];

#pragma unroll
  for (int j = 0; j < 2; ++j) {
    const size_t col  = bn + (size_t)(wn * 32 + j * 16 + lr);
    const float  bias = (float)qbias[col] * bs;
#pragma unroll
    for (int i = 0; i < 4; ++i) {
#pragma unroll
      for (int r = 0; r < 8; ++r) {
        const size_t row = bm + (size_t)(wm * 64 + i * 16 + lh * 8 + r);
        out[row * (size_t)N + col] = acc[i][j][r] * s + bias;
      }
    }
  }
}

// ------------------------------ launcher -----------------------------------

extern "C" void kernel_launch(void* const* d_in, const int* in_sizes, int n_in,
                              void* d_out, int out_size, void* d_ws, size_t ws_size,
                              hipStream_t stream) {
  const float* x      = (const float*)d_in[0];
  const int*   qw     = (const int*)d_in[1];
  const int*   qb     = (const int*)d_in[2];
  const float* scale  = (const float*)d_in[3];
  const float* bscale = (const float*)d_in[4];
  float*       out    = (float*)d_out;

  const int       D_OUT  = in_sizes[2];                           // 16384
  const long long WN     = (long long)in_sizes[1];                // D_OUT*D_IN
  const int       D_IN   = (int)(WN / D_OUT);                     // 4096
  const int       TOKENS = (int)((long long)in_sizes[0] / D_IN);  // 8192

  unsigned short* xb = (unsigned short*)d_ws;                     // 64 MB
  unsigned short* wb = xb + (size_t)TOKENS * (size_t)D_IN;        // +128 MB

  const long long nx = (long long)TOKENS * D_IN;
  const long long nw = (long long)D_OUT * D_IN;

  cvt_f32_to_bf16<<<(unsigned)((nx + 2047) / 2048), 256, 0, stream>>>(x, xb, nx);
  cvt_i32_to_bf16<<<(unsigned)((nw + 2047) / 2048), 256, 0, stream>>>(qw, wb, nw);

  dim3 grid((unsigned)(D_OUT / BN), (unsigned)(TOKENS / BM));
  qlinear_wmma_bf16<<<grid, 256, 0, stream>>>(xb, wb, qb, scale, bscale, out,
                                              TOKENS, D_OUT, D_IN);
}
